// ModelClass_14070312862196
// MI455X (gfx1250) — matrix operand order repaired and checked by reading, not compiled, not run on hardware
//
#include <hip/hip_runtime.h>

// ---------------------------------------------------------------------------
// Tree-GNN forward for MI455X (gfx1250, wave32, WMMA).
// GEMMs via v_wmma_f32_16x16x32_f16; each MLP fused into one kernel with
// inter-layer transpose through per-wave LDS; edge scatter fused into f32
// atomic stores. Activations kept as an f16 sidecar (x16, g16) so A-fragment
// loads are raw b128 loads with no f32->f16 conversion VALU work; f32 exists
// only for atomic accumulators (aggr, gsum) and the final output.
// ---------------------------------------------------------------------------

typedef __attribute__((ext_vector_type(16))) _Float16 v16h;
typedef __attribute__((ext_vector_type(8)))  _Float16 v8h;
typedef __attribute__((ext_vector_type(8)))  float    v8f;

#define NODES   1023
#define XSTRIDE (NODES * 64)   // elements per batch item in x / aggr
#define BATCH   256

static __device__ __forceinline__ v8f wmma16(v16h a, v16h b, v8f c) {
  return __builtin_amdgcn_wmma_f32_16x16x32_f16(false, a, false, b, (short)0, c,
                                                false, false);
}

// A-fragment (16x32 f16, MxK): lane holds row m = lane&15, half = lane>>4.
// e<8 -> K = half*8+e ; e>=8 -> K = 16+half*8+(e-8).
// From f16 memory: two raw 16-byte loads, no conversion.
static __device__ __forceinline__ v16h afrag_f16(const _Float16* __restrict__ p,
                                                 int hi) {
  const v8h lo = *(const v8h*)(p + hi * 8);
  const v8h up = *(const v8h*)(p + 16 + hi * 8);
  v16h a;
#pragma unroll
  for (int e = 0; e < 8; ++e) { a[e] = lo[e]; a[e + 8] = up[e]; }
  return a;
}

// From f32 memory (atomic accumulators only): convert on load.
static __device__ __forceinline__ v16h afrag_f32(const float* __restrict__ p, int hi) {
  v16h a;
#pragma unroll
  for (int e = 0; e < 8; ++e) {
    a[e]     = (_Float16)p[hi * 8 + e];
    a[e + 8] = (_Float16)p[16 + hi * 8 + e];
  }
  return a;
}

// Same A layout sourced from an LDS f16 row (p = row base + kt*32).
static __device__ __forceinline__ v16h afrag_lds(const _Float16* p, int hi) {
  v16h a;
  const _Float16* q = p + hi * 8;
#pragma unroll
  for (int e = 0; e < 8; ++e) { a[e] = q[e]; a[e + 8] = q[16 + e]; }
  return a;
}

// One hidden layer: A-frags (KT tiles) x packed-W -> bias+relu -> f16 LDS tile.
template <int KT, int NT, int STRIDE>
static __device__ __forceinline__ void layer_hidden(const v16h* A,
                                                    const _Float16* __restrict__ wp,
                                                    const float* __restrict__ bias,
                                                    _Float16* lds, int lane) {
  const int ncol = lane & 15, hi = lane >> 4;
#pragma unroll
  for (int nt = 0; nt < NT; ++nt) {
    v8f acc = {};
#pragma unroll
    for (int kt = 0; kt < KT; ++kt) {
      const v16h bf = *(const v16h*)(wp + (size_t)((kt * NT + nt) * 32 + lane) * 16);
      acc = wmma16(A[kt], bf, acc);
    }
    const float bv = bias[nt * 16 + ncol];
#pragma unroll
    for (int r = 0; r < 8; ++r) {
      float v = acc[r] + bv;
      v = v > 0.f ? v : 0.f;
      lds[(r + 8 * hi) * STRIDE + nt * 16 + ncol] = (_Float16)v;
    }
  }
  // Intra-wave LDS store->load ordering (DS ops are in-order per wave).
  asm volatile("s_wait_dscnt 0" ::: "memory");
}

template <int KT, int STRIDE>
static __device__ __forceinline__ void load_afrag_lds(v16h* A, const _Float16* lds,
                                                      int lane) {
  const int m = lane & 15, hi = lane >> 4;
#pragma unroll
  for (int kt = 0; kt < KT; ++kt)
    A[kt] = afrag_lds(lds + m * STRIDE + kt * 32, hi);
}

// ---------------------------------------------------------------------------
// glob_pre: rows = B*n_now, 64->64->64 relu, atomic-add result into gsum[b][64]
__global__ __launch_bounds__(128) void k_glob_pre(
    const _Float16* __restrict__ x16, float* __restrict__ gsum,
    const _Float16* __restrict__ w1, const float* __restrict__ b1,
    const _Float16* __restrict__ w2, const float* __restrict__ b2,
    int n_now, int nTiles) {
  __shared__ __align__(16) _Float16 lds[4][16 * 72];
  const int lane = threadIdx.x & 31, wv = threadIdx.x >> 5;
  const int tile = blockIdx.x * 4 + wv;
  if (tile >= nTiles) return;
  const int m = lane & 15, hi = lane >> 4, ncol = lane & 15;
  const unsigned row = tile * 16 + m;
  const unsigned b = row / (unsigned)n_now;
  const unsigned i = row - b * (unsigned)n_now;
  const _Float16* xr = x16 + (size_t)b * XSTRIDE + (size_t)i * 64;
  v16h A[2];
  A[0] = afrag_f16(xr, hi);
  A[1] = afrag_f16(xr + 32, hi);
  layer_hidden<2, 4, 72>(A, w1, b1, &lds[wv][0], lane);
  load_afrag_lds<2, 72>(A, &lds[wv][0], lane);
#pragma unroll
  for (int nt = 0; nt < 4; ++nt) {
    v8f acc = {};
#pragma unroll
    for (int kt = 0; kt < 2; ++kt) {
      const v16h bf = *(const v16h*)(w2 + (size_t)((kt * 4 + nt) * 32 + lane) * 16);
      acc = wmma16(A[kt], bf, acc);
    }
    const float bv = b2[nt * 16 + ncol];
#pragma unroll
    for (int r = 0; r < 8; ++r) {
      float v = acc[r] + bv;
      v = v > 0.f ? v : 0.f;
      const unsigned rr = tile * 16 + r + 8 * hi;
      const unsigned bb = rr / (unsigned)n_now;
      unsafeAtomicAdd(&gsum[bb * 64 + nt * 16 + ncol], v);
    }
  }
}

// glob_post: rows = B, 64->64->64 relu, gsum (f32) -> g16
__global__ __launch_bounds__(128) void k_glob_post(
    const float* __restrict__ gsum, _Float16* __restrict__ g16,
    const _Float16* __restrict__ w1, const float* __restrict__ b1,
    const _Float16* __restrict__ w2, const float* __restrict__ b2, int nTiles) {
  __shared__ __align__(16) _Float16 lds[4][16 * 72];
  const int lane = threadIdx.x & 31, wv = threadIdx.x >> 5;
  const int tile = blockIdx.x * 4 + wv;
  if (tile >= nTiles) return;
  const int m = lane & 15, hi = lane >> 4, ncol = lane & 15;
  const unsigned row = tile * 16 + m;
  const float* xr = gsum + (size_t)row * 64;
  v16h A[2];
  A[0] = afrag_f32(xr, hi);
  A[1] = afrag_f32(xr + 32, hi);
  layer_hidden<2, 4, 72>(A, w1, b1, &lds[wv][0], lane);
  load_afrag_lds<2, 72>(A, &lds[wv][0], lane);
#pragma unroll
  for (int nt = 0; nt < 4; ++nt) {
    v8f acc = {};
#pragma unroll
    for (int kt = 0; kt < 2; ++kt) {
      const v16h bf = *(const v16h*)(w2 + (size_t)((kt * 4 + nt) * 32 + lane) * 16);
      acc = wmma16(A[kt], bf, acc);
    }
    const float bv = b2[nt * 16 + ncol];
#pragma unroll
    for (int r = 0; r < 8; ++r) {
      float v = acc[r] + bv;
      v = v > 0.f ? v : 0.f;
      const unsigned rr = tile * 16 + r + 8 * hi;
      g16[(size_t)rr * 64 + nt * 16 + ncol] = (_Float16)v;
    }
  }
}

// proj: rows = B*2^t, (x_leaf ++ g)=128 -> 128 -> 128 -> 128 relu,
// reshaped write into the two children of each leaf (f16 sidecar only;
// f32 output is produced solely by the update kernel).
__global__ __launch_bounds__(128) void k_proj(
    _Float16* __restrict__ x16, const _Float16* __restrict__ g16,
    const _Float16* __restrict__ w1, const float* __restrict__ b1,
    const _Float16* __restrict__ w2, const float* __restrict__ b2,
    const _Float16* __restrict__ w3, const float* __restrict__ b3,
    int t, int nTiles) {
  __shared__ __align__(16) _Float16 lds[4][16 * 136];
  const int lane = threadIdx.x & 31, wv = threadIdx.x >> 5;
  const int tile = blockIdx.x * 4 + wv;
  if (tile >= nTiles) return;
  const int m = lane & 15, hi = lane >> 4, ncol = lane & 15;
  const int off0 = (1 << t) - 1, off1 = (1 << (t + 1)) - 1;
  const unsigned Lm1 = (1u << t) - 1u;
  const unsigned row = tile * 16 + m;
  const unsigned b = row >> t, i = row & Lm1;
  const _Float16* xr = x16 + (size_t)b * XSTRIDE + (size_t)(off0 + i) * 64;
  const _Float16* gr = g16 + (size_t)b * 64;
  v16h A[4];
  A[0] = afrag_f16(xr, hi);
  A[1] = afrag_f16(xr + 32, hi);
  A[2] = afrag_f16(gr, hi);
  A[3] = afrag_f16(gr + 32, hi);
  layer_hidden<4, 8, 136>(A, w1, b1, &lds[wv][0], lane);
  load_afrag_lds<4, 136>(A, &lds[wv][0], lane);
  layer_hidden<4, 8, 136>(A, w2, b2, &lds[wv][0], lane);
  load_afrag_lds<4, 136>(A, &lds[wv][0], lane);
#pragma unroll
  for (int nt = 0; nt < 8; ++nt) {
    v8f acc = {};
#pragma unroll
    for (int kt = 0; kt < 4; ++kt) {
      const v16h bf = *(const v16h*)(w3 + (size_t)((kt * 8 + nt) * 32 + lane) * 16);
      acc = wmma16(A[kt], bf, acc);
    }
    const float bv = b3[nt * 16 + ncol];
#pragma unroll
    for (int r = 0; r < 8; ++r) {
      float v = acc[r] + bv;
      v = v > 0.f ? v : 0.f;
      const unsigned rr = tile * 16 + r + 8 * hi;
      const unsigned bb = rr >> t, ii = rr & Lm1;
      // children pair is 128 contiguous elements starting at node off1+2*ii
      x16[(size_t)bb * XSTRIDE + (size_t)(off1 + 2 * ii) * 64 + nt * 16 + ncol] =
          (_Float16)v;
    }
  }
}

// msg: rows = B*E, (x[src] ++ g)=128 -> 64 -> 64 -> 64 relu,
// atomic scatter-add into aggr[b][dst] (f32).
__global__ __launch_bounds__(128) void k_msg(
    const _Float16* __restrict__ x16, const _Float16* __restrict__ g16,
    const int* __restrict__ srcIdx, const int* __restrict__ dstIdx,
    float* __restrict__ aggr,
    const _Float16* __restrict__ w1, const float* __restrict__ b1,
    const _Float16* __restrict__ w2, const float* __restrict__ b2,
    const _Float16* __restrict__ w3, const float* __restrict__ b3,
    int nE, int nTiles) {
  __shared__ __align__(16) _Float16 lds[4][16 * 72];
  const int lane = threadIdx.x & 31, wv = threadIdx.x >> 5;
  const int tile = blockIdx.x * 4 + wv;
  if (tile >= nTiles) return;
  const int m = lane & 15, hi = lane >> 4, ncol = lane & 15;
  const unsigned row = tile * 16 + m;
  const unsigned b = row / (unsigned)nE;
  const unsigned e = row - b * (unsigned)nE;
  const _Float16* xr = x16 + (size_t)b * XSTRIDE + (size_t)srcIdx[e] * 64;
  const _Float16* gr = g16 + (size_t)b * 64;
  v16h A[4];
  A[0] = afrag_f16(xr, hi);
  A[1] = afrag_f16(xr + 32, hi);
  A[2] = afrag_f16(gr, hi);
  A[3] = afrag_f16(gr + 32, hi);
  layer_hidden<4, 4, 72>(A, w1, b1, &lds[wv][0], lane);   // 128 -> 64
  load_afrag_lds<2, 72>(A, &lds[wv][0], lane);
  layer_hidden<2, 4, 72>(A, w2, b2, &lds[wv][0], lane);   // 64 -> 64
  load_afrag_lds<2, 72>(A, &lds[wv][0], lane);
#pragma unroll
  for (int nt = 0; nt < 4; ++nt) {
    v8f acc = {};
#pragma unroll
    for (int kt = 0; kt < 2; ++kt) {
      const v16h bf = *(const v16h*)(w3 + (size_t)((kt * 4 + nt) * 32 + lane) * 16);
      acc = wmma16(A[kt], bf, acc);
    }
    const float bv = b3[nt * 16 + ncol];
#pragma unroll
    for (int r = 0; r < 8; ++r) {
      float v = acc[r] + bv;
      v = v > 0.f ? v : 0.f;
      const unsigned rr = tile * 16 + r + 8 * hi;
      const unsigned bb = rr / (unsigned)nE;
      const unsigned ee = rr - bb * (unsigned)nE;
      const int d = dstIdx[ee];
      unsafeAtomicAdd(&aggr[(size_t)bb * XSTRIDE + (size_t)d * 64 + nt * 16 + ncol], v);
    }
  }
}

// update: rows = B*n_next, (x ++ aggr ++ g)=192 -> 192 -> 64 -> 64 relu,
// overwrites x16 in place (each row reads only itself) and writes the f32
// output (final step's pass covers the whole d_out).
__global__ __launch_bounds__(128) void k_update(
    _Float16* __restrict__ x16, const float* __restrict__ aggr,
    const _Float16* __restrict__ g16, float* __restrict__ x_out,
    const _Float16* __restrict__ w1, const float* __restrict__ b1,
    const _Float16* __restrict__ w2, const float* __restrict__ b2,
    const _Float16* __restrict__ w3, const float* __restrict__ b3,
    int n_next, int nTiles) {
  __shared__ __align__(16) _Float16 lds[4][16 * 200];
  const int lane = threadIdx.x & 31, wv = threadIdx.x >> 5;
  const int tile = blockIdx.x * 4 + wv;
  if (tile >= nTiles) return;
  const int m = lane & 15, hi = lane >> 4, ncol = lane & 15;
  const unsigned row = tile * 16 + m;
  const unsigned b = row / (unsigned)n_next;
  const unsigned i = row - b * (unsigned)n_next;
  const _Float16* xr = x16 + (size_t)b * XSTRIDE + (size_t)i * 64;
  const float* ar = aggr + (size_t)b * XSTRIDE + (size_t)i * 64;
  const _Float16* gr = g16 + (size_t)b * 64;
  v16h A[6];
  A[0] = afrag_f16(xr, hi);
  A[1] = afrag_f16(xr + 32, hi);
  A[2] = afrag_f32(ar, hi);
  A[3] = afrag_f32(ar + 32, hi);
  A[4] = afrag_f16(gr, hi);
  A[5] = afrag_f16(gr + 32, hi);
  layer_hidden<6, 12, 200>(A, w1, b1, &lds[wv][0], lane);  // 192 -> 192
  load_afrag_lds<6, 200>(A, &lds[wv][0], lane);
  layer_hidden<6, 4, 200>(A, w2, b2, &lds[wv][0], lane);   // 192 -> 64
  load_afrag_lds<2, 200>(A, &lds[wv][0], lane);
#pragma unroll
  for (int nt = 0; nt < 4; ++nt) {
    v8f acc = {};
#pragma unroll
    for (int kt = 0; kt < 2; ++kt) {
      const v16h bf = *(const v16h*)(w3 + (size_t)((kt * 4 + nt) * 32 + lane) * 16);
      acc = wmma16(A[kt], bf, acc);
    }
    const float bv = b3[nt * 16 + ncol];
#pragma unroll
    for (int r = 0; r < 8; ++r) {
      float v = acc[r] + bv;
      v = v > 0.f ? v : 0.f;
      const unsigned rr = tile * 16 + r + 8 * hi;
      const unsigned bb = rr / (unsigned)n_next;
      const unsigned ii = rr - bb * (unsigned)n_next;
      const size_t idx = (size_t)bb * XSTRIDE + (size_t)ii * 64 + nt * 16 + ncol;
      x_out[idx] = v;
      x16[idx] = (_Float16)v;
    }
  }
}

// ---------------------------------------------------------------------------
// Utility kernels
__global__ void k_zero(float* p, long n) {
  long id = (long)blockIdx.x * blockDim.x + threadIdx.x;
  if (id < n) p[id] = 0.f;
}

// Pack f32 row-major W[K][N] into WMMA B-operand tile layout (f16):
// tile(kt,nt) -> 32 lanes x 16 halves; lane: n = nt*16+(lane&15),
// element e: k = kt*32 + (e>=8)*16 + (lane>>4)*8 + (e&7).
__global__ void k_pack(const float* __restrict__ W, _Float16* __restrict__ out,
                       int N, int total) {
  int id = blockIdx.x * blockDim.x + threadIdx.x;
  if (id >= total) return;
  const int e = id & 15, lane = (id >> 4) & 31, tile = id >> 9;
  const int NT = N >> 4;
  const int nt = tile % NT, kt = tile / NT;
  const int k = kt * 32 + (e & 8) * 2 + (lane >> 4) * 8 + (e & 7);
  const int n = nt * 16 + (lane & 15);
  out[id] = (_Float16)W[(size_t)k * N + n];
}

// Deterministic tree edge lists (grouped by dst level ascending, matching the
// reference; edges for step t are the first E_t = t*2^(t+2)+2 entries).
__global__ void k_edges(int* __restrict__ src, int* __restrict__ dst) {
  const int d = threadIdx.x;
  if (d < 1 || d >= NODES) return;
  const int l = 31 - __clz(d + 1);
  const int i = (d + 1) - (1 << l);
  const int base = (l >= 2) ? ((l - 2) * (1 << l) + 2) : 0;
  int eo = base + i * l;
  int al = l, ai = i;
  while (al > 0) {
    al--;
    ai >>= 1;
    src[eo] = (1 << al) - 1 + ai;
    dst[eo] = d;
    eo++;
  }
}

__global__ void k_init_x(const float* __restrict__ rv, _Float16* __restrict__ x16) {
  int id = blockIdx.x * blockDim.x + threadIdx.x;
  if (id >= BATCH * 64) return;
  x16[(size_t)(id >> 6) * XSTRIDE + (id & 63)] = (_Float16)rv[id];
}

// ---------------------------------------------------------------------------
extern "C" void kernel_launch(void* const* d_in, const int* in_sizes, int n_in,
                              void* d_out, int out_size, void* d_ws, size_t ws_size,
                              hipStream_t stream) {
  (void)in_sizes; (void)n_in; (void)out_size; (void)ws_size;

  const float* rv = (const float*)d_in[0];
  float* x_out = (float*)d_out;  // final output (256*1023*64 f32)

  char* ws = (char*)d_ws;
  float*    aggr   = (float*)ws;                       // 256*65472 f32
  float*    gsum   = (float*)(ws + 67043328);          // 256*64 f32
  _Float16* g16    = (_Float16*)(ws + 67108864);       // 256*64 f16
  int*      srcIdx = (int*)(ws + 67141632);            // 8194 ints (padded)
  int*      dstIdx = (int*)(ws + 67174656);            // 8194 ints (padded)
  _Float16* wp     = (_Float16*)(ws + 67207680);       // packed weights (f16)
  _Float16* x16    = (_Float16*)(ws + 67478016);       // 256*65472 f16

  // d_in layout: rv, then (W,b) pairs in dict insertion order.
  struct LayerDesc { int idx, K, N; };
  const LayerDesc L[13] = {
      {1, 64, 64},   {3, 64, 64},                       // glob_pre
      {5, 64, 64},   {7, 64, 64},                       // glob_post
      {9, 128, 128}, {11, 128, 128}, {13, 128, 128},    // proj
      {15, 128, 64}, {17, 64, 64},   {19, 64, 64},      // msg
      {21, 192, 192},{23, 192, 64},  {25, 64, 64}       // update
  };
  _Float16* wofs[13];
  const float* bptr[13];
  size_t o = 0;
  for (int j = 0; j < 13; ++j) {
    wofs[j] = wp + o;
    bptr[j] = (const float*)d_in[L[j].idx + 1];
    const int tot = L[j].K * L[j].N;
    k_pack<<<(tot + 255) / 256, 256, 0, stream>>>((const float*)d_in[L[j].idx],
                                                  wofs[j], L[j].N, tot);
    o += (size_t)tot;
  }
  k_edges<<<1, 1024, 0, stream>>>(srcIdx, dstIdx);
  k_init_x<<<(BATCH * 64 + 255) / 256, 256, 0, stream>>>(rv, x16);

  for (int t = 0; t < 9; ++t) {
    const int n_now = (1 << (t + 1)) - 1;
    const int n_next = (1 << (t + 2)) - 1;
    const int E = t * (1 << (t + 2)) + 2;

    // aggr + gsum are adjacent: zero both in one launch (16,777,216 floats)
    const long zn = (long)BATCH * XSTRIDE + (long)BATCH * 64;
    k_zero<<<(int)((zn + 511) / 512), 512, 0, stream>>>(aggr, zn);

    int tiles = BATCH * n_now / 16;
    k_glob_pre<<<(tiles + 3) / 4, 128, 0, stream>>>(
        x16, gsum, wofs[0], bptr[0], wofs[1], bptr[1], n_now, tiles);

    tiles = BATCH / 16;
    k_glob_post<<<(tiles + 3) / 4, 128, 0, stream>>>(
        gsum, g16, wofs[2], bptr[2], wofs[3], bptr[3], tiles);

    tiles = (BATCH << t) / 16;
    k_proj<<<(tiles + 3) / 4, 128, 0, stream>>>(
        x16, g16, wofs[4], bptr[4], wofs[5], bptr[5], wofs[6], bptr[6], t, tiles);

    tiles = BATCH * E / 16;
    k_msg<<<(tiles + 3) / 4, 128, 0, stream>>>(
        x16, g16, srcIdx, dstIdx, aggr, wofs[7], bptr[7], wofs[8], bptr[8],
        wofs[9], bptr[9], E, tiles);

    tiles = BATCH * n_next / 16;
    k_update<<<(tiles + 3) / 4, 128, 0, stream>>>(
        x16, aggr, g16, x_out, wofs[10], bptr[10], wofs[11], bptr[11], wofs[12],
        bptr[12], n_next, tiles);
  }
}
